// gmm_mil_17394617549084
// MI455X (gfx1250) — matrix operand order repaired and checked by reading, not compiled, not run on hardware
//
#include <hip/hip_runtime.h>
#include <hip/hip_bf16.h>

#define B_    8
#define N_    2048
#define F_    1024
#define Z_    256
#define K_    10
#define NOUT_ 2
#define DPOOL_ (K_*Z_ + K_*Z_*Z_)
#define REG_  1e-6f
#define LOG2PI_ 1.8378770664093453f

typedef __attribute__((ext_vector_type(16))) _Float16 v16h;
typedef __attribute__((ext_vector_type(8)))  float    v8f;

// Build a 16x32 f16 A-fragment (ISA 7.12.2 layout) from a row-major fp32 row.
// Per lane the two 8-element K runs are contiguous -> two pairs of float4 loads.
__device__ __forceinline__ v16h load_a_frag_f32row(const float* __restrict__ rowp,
                                                   int k0, int h) {
    const float4* p0 = reinterpret_cast<const float4*>(rowp + k0 + 8*h);
    const float4* p1 = reinterpret_cast<const float4*>(rowp + k0 + 8*h + 16);
    float4 a0 = p0[0], a1 = p0[1], a2 = p1[0], a3 = p1[1];
    v16h f;
    f[0]  = (_Float16)a0.x; f[1]  = (_Float16)a0.y; f[2]  = (_Float16)a0.z; f[3]  = (_Float16)a0.w;
    f[4]  = (_Float16)a1.x; f[5]  = (_Float16)a1.y; f[6]  = (_Float16)a1.z; f[7]  = (_Float16)a1.w;
    f[8]  = (_Float16)a2.x; f[9]  = (_Float16)a2.y; f[10] = (_Float16)a2.z; f[11] = (_Float16)a2.w;
    f[12] = (_Float16)a3.x; f[13] = (_Float16)a3.y; f[14] = (_Float16)a3.z; f[15] = (_Float16)a3.w;
    return f;
}

// ---------------------------------------------------------------------------
// Encoder: emb[b,n,z] = relu(bags[b,n,:] @ Wenc[:,z] + benc[z])
// One wave per 64x16 output strip: B fragment reused by 4 row tiles ->
// 4x v_wmma_f32_16x16x32_f16 per 32-deep K step.
// ---------------------------------------------------------------------------
__global__ void encode_wmma(const float* __restrict__ bags,
                            const float* __restrict__ Wenc,
                            const float* __restrict__ benc,
                            float* __restrict__ emb) {
    int tile = blockIdx.x;
    int zt = tile % (Z_/16); tile /= (Z_/16);
    int nt = tile % (N_/64);
    int b  = tile / (N_/64);
    int lane = threadIdx.x & 31;
    int m = lane & 15;
    int h = lane >> 4;
    int row0 = nt * 64, col0 = zt * 16;
    const float* A0 = bags + ((size_t)b * N_ + row0 + m) * F_;

    v8f acc[4] = {{}, {}, {}, {}};
    for (int k0 = 0; k0 < F_; k0 += 32) {
        if (k0 + 32 < F_) {
            __builtin_prefetch(A0 + k0 + 32, 0, 1);                    // global_prefetch_b8
            __builtin_prefetch(Wenc + (size_t)(k0 + 32 + 16*h) * Z_ + col0, 0, 1);
        }
        v16h bfrag;
#pragma unroll
        for (int e = 0; e < 16; ++e)
            bfrag[e] = (_Float16)Wenc[(size_t)(k0 + 8*h + e + (e >= 8 ? 8 : 0)) * Z_ + col0 + m];
#pragma unroll
        for (int t = 0; t < 4; ++t) {
            v16h afrag = load_a_frag_f32row(A0 + (size_t)t * 16 * F_, k0, h);
            acc[t] = __builtin_amdgcn_wmma_f32_16x16x32_f16(
                false, afrag, false, bfrag, (short)0, acc[t], false, false);
        }
    }
    float bias = benc[col0 + m];
#pragma unroll
    for (int t = 0; t < 4; ++t) {
#pragma unroll
        for (int p = 0; p < 8; ++p) {
            int mrow = row0 + t*16 + 8*h + p;   // D layout: VGPR p -> row 8h+p
            float v = acc[t][p] + bias;
            v = v > 0.f ? v : 0.f;
            emb[((size_t)b * N_ + mrow) * Z_ + col0 + m] = v;
        }
    }
}

// ---------------------------------------------------------------------------
// Per-(bag,dim) variance of emb over N (for diagonal covariance init)
// ---------------------------------------------------------------------------
__global__ void colstats(const float* __restrict__ emb, float* __restrict__ varz) {
    int t = blockIdx.x * blockDim.x + threadIdx.x;
    if (t >= B_ * Z_) return;
    int b = t / Z_, z = t % Z_;
    float s = 0.f, ss = 0.f;
    for (int n = 0; n < N_; ++n) {
        float v = emb[((size_t)b * N_ + n) * Z_ + z];
        s += v; ss += v * v;
    }
    float mu = s / (float)N_;
    varz[t] = ss / (float)N_ - mu * mu;
}

__global__ void init_means_logw(const float* __restrict__ emb,
                                float* __restrict__ means,
                                float* __restrict__ logw) {
    int t = blockIdx.x * blockDim.x + threadIdx.x;
    if (t >= B_ * K_ * Z_) return;
    int z = t % Z_;
    int k = (t / Z_) % K_;
    int b = t / (Z_ * K_);
    int idx = (int)((float)(N_ - 1) * (float)k / (float)(K_ - 1));
    means[t] = emb[((size_t)b * N_ + idx) * Z_ + z];
    if (z == 0) logw[b * K_ + k] = -logf((float)K_);
}

__global__ void init_covs(const float* __restrict__ varz, float* __restrict__ covs) {
    size_t t = (size_t)blockIdx.x * blockDim.x + threadIdx.x;
    if (t >= (size_t)B_ * K_ * Z_ * Z_) return;
    int e = (int)(t % Z_);
    int z = (int)((t / Z_) % Z_);
    int b = (int)(t / ((size_t)Z_ * Z_ * K_));
    covs[t] = (z == e) ? (varz[b * Z_ + z] + REG_) : 0.f;
}

// ---------------------------------------------------------------------------
// In-place Cholesky of covs[bk] (lower), one workgroup per (b,k). Z_ threads,
// thread = row. Also produces logdet = 2*sum(log diag(L)).
// ---------------------------------------------------------------------------
__global__ void cholesky(float* __restrict__ covs, float* __restrict__ logdet) {
    int bk = blockIdx.x;
    float* A = covs + (size_t)bk * Z_ * Z_;
    int tid = threadIdx.x;         // row index
    for (int j = tid + 1; j < Z_; ++j) A[(size_t)tid * Z_ + j] = 0.f;
    __syncthreads();
    for (int j = 0; j < Z_; ++j) {
        if (tid == j) A[(size_t)j * Z_ + j] = sqrtf(fmaxf(A[(size_t)j * Z_ + j], 1e-20f));
        __syncthreads();
        float Ljj = A[(size_t)j * Z_ + j];
        if (tid > j) A[(size_t)tid * Z_ + j] /= Ljj;
        __syncthreads();
        if (tid > j) {
            float Lij = A[(size_t)tid * Z_ + j];
            for (int t = j + 1; t <= tid; ++t)
                A[(size_t)tid * Z_ + t] -= Lij * A[(size_t)t * Z_ + j];
        }
        __syncthreads();
    }
    __shared__ float red[Z_];
    red[tid] = logf(fmaxf(A[(size_t)tid * Z_ + tid], 1e-30f));
    __syncthreads();
    for (int s = Z_ / 2; s > 0; s >>= 1) {
        if (tid < s) red[tid] += red[tid + s];
        __syncthreads();
    }
    if (tid == 0) logdet[bk] = 2.f * red[0];
}

// ---------------------------------------------------------------------------
// Log-prob via forward substitution; one point per thread (y[256] in scratch).
// ---------------------------------------------------------------------------
__global__ void logprob(const float* __restrict__ emb, const float* __restrict__ L_,
                        const float* __restrict__ means, const float* __restrict__ logdet,
                        const float* __restrict__ logw, float* __restrict__ r) {
    int blk = blockIdx.x;
    int nc = blk % (N_ / 64); blk /= (N_ / 64);
    int k = blk % K_;
    int b = blk / K_;
    int n = nc * 64 + threadIdx.x;
    const float* L  = L_    + (size_t)(b * K_ + k) * Z_ * Z_;
    const float* mu = means + (size_t)(b * K_ + k) * Z_;
    const float* x  = emb   + ((size_t)b * N_ + n) * Z_;
    float y[Z_];
    float maha = 0.f;
    for (int i = 0; i < Z_; ++i) {
        float s = x[i] - mu[i];
        const float* Li = L + (size_t)i * Z_;
        for (int j = 0; j < i; ++j) s -= Li[j] * y[j];
        float yi = s / Li[i];
        y[i] = yi;
        maha += yi * yi;
    }
    float lp = -0.5f * ((float)Z_ * LOG2PI_ + logdet[b * K_ + k] + maha);
    r[((size_t)b * N_ + n) * K_ + k] = lp + logw[b * K_ + k];
}

__global__ void softmax_rows(float* __restrict__ r) {
    int t = blockIdx.x * blockDim.x + threadIdx.x;
    if (t >= B_ * N_) return;
    float* row = r + (size_t)t * K_;
    float mx = row[0];
    for (int k = 1; k < K_; ++k) mx = fmaxf(mx, row[k]);
    float sum = 0.f;
    for (int k = 0; k < K_; ++k) sum += expf(row[k] - mx);
    float lse = mx + logf(sum);
    for (int k = 0; k < K_; ++k) row[k] = expf(row[k] - lse);
}

__global__ void nk_kernel(const float* __restrict__ r, float* __restrict__ Nk) {
    int bk = blockIdx.x;
    int b = bk / K_, k = bk % K_;
    int tid = threadIdx.x;
    float s = 0.f;
    for (int n = tid; n < N_; n += blockDim.x)
        s += r[((size_t)b * N_ + n) * K_ + k];
    __shared__ float red[256];
    red[tid] = s;
    __syncthreads();
    for (int st = 128; st > 0; st >>= 1) {
        if (tid < st) red[tid] += red[tid + st];
        __syncthreads();
    }
    if (tid == 0) Nk[bk] = red[0] + 1e-10f;
}

__global__ void means_kernel(const float* __restrict__ r, const float* __restrict__ emb,
                             const float* __restrict__ Nk, float* __restrict__ means) {
    int bk = blockIdx.x;
    int b = bk / K_, k = bk % K_;
    int z = threadIdx.x;
    float s = 0.f;
    for (int n = 0; n < N_; ++n)
        s += r[((size_t)b * N_ + n) * K_ + k] * emb[((size_t)b * N_ + n) * Z_ + z];
    means[(size_t)bk * Z_ + z] = s / Nk[bk];
}

// ---------------------------------------------------------------------------
// M-step covariance: cov[bk] = (A_wᵀ A)/Nk + reg*I,  A[n,d]=diff, A_w[n,d]=r*diff.
// One wave per 32x32 output tile (2x2 register blocking) -> 4 WMMA per N-step.
// ---------------------------------------------------------------------------
__global__ void cov_wmma(const float* __restrict__ r, const float* __restrict__ emb,
                         const float* __restrict__ means, const float* __restrict__ Nk,
                         float* __restrict__ covs) {
    int tile = blockIdx.x;
    int ct = tile % (Z_/32); tile /= (Z_/32);
    int rt = tile % (Z_/32); tile /= (Z_/32);
    int k  = tile % K_;
    int b  = tile / K_;
    int bk = b * K_ + k;
    int lane = threadIdx.x & 31;
    int m = lane & 15, h = lane >> 4;
    int row0 = rt * 32, col0 = ct * 32;
    const float* mu = means + (size_t)bk * Z_;
    float muA0 = mu[row0 + m],      muA1 = mu[row0 + 16 + m];
    float muB0 = mu[col0 + m],      muB1 = mu[col0 + 16 + m];
    const float* rb = r   + (size_t)b * N_ * K_ + k;
    const float* eb = emb + (size_t)b * N_ * Z_;

    v8f acc00 = {}, acc01 = {}, acc10 = {}, acc11 = {};
    for (int n0 = 0; n0 < N_; n0 += 32) {
        if (n0 + 32 < N_)
            __builtin_prefetch(eb + (size_t)(n0 + 32 + 8*h) * Z_ + row0, 0, 1);
        float rv[16];
#pragma unroll
        for (int e = 0; e < 16; ++e) {
            int na = n0 + 8*h + e + (e >= 8 ? 8 : 0);
            rv[e] = rb[(size_t)na * K_];
        }
        v16h a0, a1, b0, b1;
#pragma unroll
        for (int e = 0; e < 16; ++e) {
            int na = n0 + 8*h + e + (e >= 8 ? 8 : 0);     // A-fragment K index
            const float* ea = eb + (size_t)na * Z_;
            a0[e] = (_Float16)(rv[e] * (ea[row0 + m]      - muA0));
            a1[e] = (_Float16)(rv[e] * (ea[row0 + 16 + m] - muA1));
            int nb = n0 + 8*h + e + (e >= 8 ? 8 : 0);     // B-fragment K index
            const float* ebp = eb + (size_t)nb * Z_;
            b0[e] = (_Float16)(ebp[col0 + m]      - muB0);
            b1[e] = (_Float16)(ebp[col0 + 16 + m] - muB1);
        }
        acc00 = __builtin_amdgcn_wmma_f32_16x16x32_f16(false, a0, false, b0, (short)0, acc00, false, false);
        acc01 = __builtin_amdgcn_wmma_f32_16x16x32_f16(false, a0, false, b1, (short)0, acc01, false, false);
        acc10 = __builtin_amdgcn_wmma_f32_16x16x32_f16(false, a1, false, b0, (short)0, acc10, false, false);
        acc11 = __builtin_amdgcn_wmma_f32_16x16x32_f16(false, a1, false, b1, (short)0, acc11, false, false);
    }
    float invNk = 1.f / Nk[bk];
    float* C = covs + (size_t)bk * Z_ * Z_;
#pragma unroll
    for (int p = 0; p < 8; ++p) {
        int rA = row0 + 8*h + p, rB = row0 + 16 + 8*h + p;
        int c0 = col0 + m,       c1 = col0 + 16 + m;
        float v;
        v = acc00[p] * invNk + ((rA == c0) ? REG_ : 0.f); C[(size_t)rA * Z_ + c0] = v;
        v = acc01[p] * invNk + ((rA == c1) ? REG_ : 0.f); C[(size_t)rA * Z_ + c1] = v;
        v = acc10[p] * invNk + ((rB == c0) ? REG_ : 0.f); C[(size_t)rB * Z_ + c0] = v;
        v = acc11[p] * invNk + ((rB == c1) ? REG_ : 0.f); C[(size_t)rB * Z_ + c1] = v;
    }
}

__global__ void logw_kernel(const float* __restrict__ Nk, float* __restrict__ logw) {
    int b = blockIdx.x;
    float tot = 0.f;
    for (int k = 0; k < K_; ++k) tot += Nk[b * K_ + k];
    if (threadIdx.x < K_)
        logw[b * K_ + threadIdx.x] = logf(Nk[b * K_ + threadIdx.x] / tot);
}

// ---------------------------------------------------------------------------
// Head: batch-norm over B on pooled=[means|covs], then Linear to n_out.
// ---------------------------------------------------------------------------
__device__ __forceinline__ float pooled_at(const float* means, const float* covs,
                                           int b, int d) {
    return (d < K_ * Z_) ? means[(size_t)b * K_ * Z_ + d]
                         : covs[(size_t)b * K_ * Z_ * Z_ + (d - K_ * Z_)];
}

__global__ void bnstats(const float* __restrict__ means, const float* __restrict__ covs,
                        float* __restrict__ bn_mu, float* __restrict__ bn_var) {
    int d = blockIdx.x * blockDim.x + threadIdx.x;
    if (d >= DPOOL_) return;
    float s = 0.f, ss = 0.f;
    for (int b = 0; b < B_; ++b) {
        float v = pooled_at(means, covs, b, d);
        s += v; ss += v * v;
    }
    float mu = s / (float)B_;
    bn_mu[d]  = mu;
    bn_var[d] = ss / (float)B_ - mu * mu;
}

__global__ void head_kernel(const float* __restrict__ means, const float* __restrict__ covs,
                            const float* __restrict__ bn_mu, const float* __restrict__ bn_var,
                            const float* __restrict__ gamma, const float* __restrict__ beta,
                            const float* __restrict__ Whead, const float* __restrict__ bhead,
                            float* __restrict__ out) {
    int b = blockIdx.x / NOUT_;
    int o = blockIdx.x % NOUT_;
    int tid = threadIdx.x;
    float s = 0.f;
    for (int d = tid; d < DPOOL_; d += blockDim.x) {
        float v = pooled_at(means, covs, b, d);
        float hn = (v - bn_mu[d]) * rsqrtf(bn_var[d] + 1e-5f) * gamma[d] + beta[d];
        s += hn * Whead[(size_t)d * NOUT_ + o];
    }
    __shared__ float red[256];
    red[tid] = s;
    __syncthreads();
    for (int st = 128; st > 0; st >>= 1) {
        if (tid < st) red[tid] += red[tid + st];
        __syncthreads();
    }
    if (tid == 0) out[b * NOUT_ + o] = red[0] + bhead[o];
}

// ---------------------------------------------------------------------------
extern "C" void kernel_launch(void* const* d_in, const int* in_sizes, int n_in,
                              void* d_out, int out_size, void* d_ws, size_t ws_size,
                              hipStream_t stream) {
    (void)in_sizes; (void)n_in; (void)out_size; (void)ws_size;
    const float* bags  = (const float*)d_in[0];
    const float* Wenc  = (const float*)d_in[1];
    const float* benc  = (const float*)d_in[2];
    const float* gamma = (const float*)d_in[3];
    const float* beta  = (const float*)d_in[4];
    const float* Whead = (const float*)d_in[5];
    const float* bhead = (const float*)d_in[6];
    float* out = (float*)d_out;

    float* ws = (float*)d_ws;
    float* emb    = ws; ws += (size_t)B_ * N_ * Z_;
    float* covs   = ws; ws += (size_t)B_ * K_ * Z_ * Z_;
    float* means  = ws; ws += (size_t)B_ * K_ * Z_;
    float* rbuf   = ws; ws += (size_t)B_ * N_ * K_;
    float* Nk     = ws; ws += B_ * K_;
    float* logw   = ws; ws += B_ * K_;
    float* logdet = ws; ws += B_ * K_;
    float* varz   = ws; ws += B_ * Z_;
    float* bn_mu  = ws; ws += DPOOL_;
    float* bn_var = ws; ws += DPOOL_;

    encode_wmma<<<dim3(B_ * (N_/64) * (Z_/16)), 32, 0, stream>>>(bags, Wenc, benc, emb);
    colstats<<<dim3((B_ * Z_ + 255) / 256), 256, 0, stream>>>(emb, varz);
    init_means_logw<<<dim3((B_ * K_ * Z_ + 255) / 256), 256, 0, stream>>>(emb, means, logw);
    init_covs<<<dim3((unsigned)(((size_t)B_ * K_ * Z_ * Z_ + 255) / 256)), 256, 0, stream>>>(varz, covs);

    for (int it = 0; it < 10; ++it) {
        cholesky<<<dim3(B_ * K_), Z_, 0, stream>>>(covs, logdet);
        logprob<<<dim3(B_ * K_ * (N_ / 64)), 64, 0, stream>>>(emb, covs, means, logdet, logw, rbuf);
        softmax_rows<<<dim3((B_ * N_ + 255) / 256), 256, 0, stream>>>(rbuf);
        nk_kernel<<<dim3(B_ * K_), 256, 0, stream>>>(rbuf, Nk);
        means_kernel<<<dim3(B_ * K_), Z_, 0, stream>>>(rbuf, emb, Nk, means);
        cov_wmma<<<dim3(B_ * K_ * (Z_/32) * (Z_/32)), 32, 0, stream>>>(rbuf, emb, means, Nk, covs);
        logw_kernel<<<dim3(B_), 32, 0, stream>>>(Nk, logw);
    }

    bnstats<<<dim3((DPOOL_ + 255) / 256), 256, 0, stream>>>(means, covs, bn_mu, bn_var);
    head_kernel<<<dim3(B_ * NOUT_), 256, 0, stream>>>(means, covs, bn_mu, bn_var,
                                                      gamma, beta, Whead, bhead, out);
}
